// GSTRGCT_44461501449018
// MI455X (gfx1250) — matrix operand order repaired and checked by reading, not compiled, not run on hardware
//
#include <hip/hip_runtime.h>
#include <math.h>

// ---------------- types for WMMA ----------------
typedef __attribute__((ext_vector_type(16))) _Float16 v16h;
typedef __attribute__((ext_vector_type(8)))  _Float16 v8h;
typedef __attribute__((ext_vector_type(8)))  float    v8f;

#define NB   512     // nodes
#define BB   32      // batch
#define TIN  24
#define TOUT 12
#define CO   64
#define DM   64
#define DFF  256
#define SEQ  (BB*NB)           // 16384 series
#define ROWS ((long)SEQ*TIN)   // 393216 GEMM rows for temporal branch

// =========================================================================
// Generic batched GEMM:  C[b] = act( A[b](MxK) * Bt[b](NxK)^T + bias )
// A, Bt are f16 row-major (Bt holds B transposed -> contiguous K per row).
// One wave computes a 16 x (16*NT) strip: the A fragment is loaded once per
// k-step and reused by NT independent v_wmma_f32_16x16x32_f16 chains, which
// amortizes VMEM traffic and gives the XDL pipe back-to-back WMMAs.
// A/B fragments: contiguous K per lane half (k = 16*hi + s), same order in both (wave32).
// =========================================================================
template <int NT>
__global__ __launch_bounds__(128)
void k_gemm(const _Float16* __restrict__ A, const _Float16* __restrict__ Bt,
            const float* __restrict__ bias,
            float* __restrict__ Cf, _Float16* __restrict__ Ch,
            int M, int K, int Nstore, int ldc,
            long sA, long sB, long sC,
            int tilesN, int doRelu, int doAcc)
{
    const int wave = threadIdx.x >> 5;
    const int lane = threadIdx.x & 31;
    const int tilesNT = tilesN / NT;           // strips per row of tiles
    const int strip = blockIdx.x * 4 + wave;
    const int tilesM = M >> 4;
    if (strip >= tilesM * tilesNT) return;
    const int tm  = strip / tilesNT;
    const int tn0 = (strip - tm * tilesNT) * NT;
    const long batch = blockIdx.y;
    const int hi = lane >> 4;           // lane group (0: lanes 0-15, 1: lanes 16-31)
    const int lo = lane & 15;

    const _Float16* a_row = A  + batch * sA + (long)(tm * 16 + lo) * K;
    const _Float16* b_row = Bt + batch * sB + (long)(tn0 * 16 + lo) * K;

    v8f acc[NT];
    #pragma unroll
    for (int j = 0; j < NT; ++j) acc[j] = (v8f){};

    for (int k = 0; k < K; k += 32) {
        // A fragment: lanes 0-15 -> K 0..15; lanes 16-31 -> K 16..31
        v16h av = *(const v16h*)(a_row + k + hi * 16);
        #pragma unroll
        for (int j = 0; j < NT; ++j) {
            // B fragment: lanes 0-15 -> K 0..15 of col n; lanes 16-31 -> K 16..31
            v16h bv = *(const v16h*)(b_row + (long)j * 16 * K + k + hi * 16);
            acc[j] = __builtin_amdgcn_wmma_f32_16x16x32_f16(
                         false, av, false, bv, (short)0, acc[j], false, false);
        }
    }

    #pragma unroll
    for (int j = 0; j < NT; ++j) {
        const int col = (tn0 + j) * 16 + lo;
        if (col < Nstore) {
            const float bv = bias ? bias[col] : 0.f;
            #pragma unroll
            for (int r = 0; r < 8; ++r) {
                const int row = tm * 16 + hi * 8 + r;   // C/D layout: lanes16-31 -> M+8
                float v = acc[j][r] + bv;
                if (doRelu) v = fmaxf(v, 0.f);
                const long idx = batch * sC + (long)row * ldc + col;
                if (doAcc)      Cf[idx] += v;
                else if (Cf)    Cf[idx]  = v;
                if (Ch)         Ch[idx]  = (_Float16)v;
            }
        }
    }
}

// ---------------- small utility kernels ----------------
__global__ void k_cvt(const float* __restrict__ s, _Float16* __restrict__ d, long n) {
    long i = (long)blockIdx.x * blockDim.x + threadIdx.x;
    if (i < n) d[i] = (_Float16)s[i];
}
__global__ void k_fill0h(_Float16* __restrict__ d, long n) {
    long i = (long)blockIdx.x * blockDim.x + threadIdx.x;
    if (i < n) d[i] = (_Float16)0.f;
}

// row softmax over 512 columns (W = softmax(logits, axis=1)), in place
__global__ __launch_bounds__(256)
void k_softmax512(float* __restrict__ W) {
    __shared__ float red[256];
    float* p = W + (long)blockIdx.x * NB;
    int t = threadIdx.x;
    float a = p[t], b = p[t + 256];
    red[t] = fmaxf(a, b); __syncthreads();
    for (int off = 128; off > 0; off >>= 1) { if (t < off) red[t] = fmaxf(red[t], red[t + off]); __syncthreads(); }
    float mx = red[0]; __syncthreads();
    float e0 = __expf(a - mx), e1 = __expf(b - mx);
    red[t] = e0 + e1; __syncthreads();
    for (int off = 128; off > 0; off >>= 1) { if (t < off) red[t] += red[t + off]; __syncthreads(); }
    float inv = 1.f / red[0];
    p[t] = e0 * inv; p[t + 256] = e1 * inv;
}

// power iteration for dominant eigenvalue of L = I - W (rows of W sum to 1)
__global__ __launch_bounds__(512)
void k_power(const float* __restrict__ W, float* __restrict__ lam) {
    __shared__ float v[NB];
    __shared__ float red[NB];
    int i = threadIdx.x;
    v[i] = 0.044194174f;            // 1/sqrt(512)
    __syncthreads();
    const float* row = W + (long)i * NB;
    for (int it = 0; it < 40; ++it) {
        float s = 0.f;
        for (int j = 0; j < NB; ++j) s += row[j] * v[j];
        float yi = v[i] - s;        // (I - W) v
        red[i] = yi * yi; __syncthreads();
        for (int off = 256; off > 0; off >>= 1) { if (i < off) red[i] += red[i + off]; __syncthreads(); }
        float inv = 1.f / sqrtf(red[0] + 1e-20f);
        __syncthreads();
        v[i] = yi * inv;
        __syncthreads();
    }
    float s = 0.f;
    for (int j = 0; j < NB; ++j) s += row[j] * v[j];
    red[i] = v[i] * (v[i] - s);     // Rayleigh quotient
    __syncthreads();
    for (int off = 256; off > 0; off >>= 1) { if (i < off) red[i] += red[i + off]; __syncthreads(); }
    if (i == 0) lam[0] = red[0];
}

// Lt = 2(I-W)/lam - I ; cheb2 = 2*Lt*Lt - I (elementwise, as in source).
// Emit transposed f16 copies (chebT[m,n] = cheb[n,m]) plus W^T f16.
__global__ void k_build(const float* __restrict__ W, const float* __restrict__ lam,
                        _Float16* __restrict__ WT, _Float16* __restrict__ C1T,
                        _Float16* __restrict__ C2T) {
    long idx = (long)blockIdx.x * blockDim.x + threadIdx.x;
    if (idx >= (long)NB * NB) return;
    int i = (int)(idx >> 9), j = (int)(idx & 511);
    float w = W[idx];
    float dg = (i == j) ? 1.f : 0.f;
    float lt = 2.f * (dg - w) / lam[0] - dg;
    float c2 = 2.f * lt * lt - dg;
    long tid = (long)j * NB + i;
    WT[tid]  = (_Float16)w;
    C1T[tid] = (_Float16)lt;
    C2T[tid] = (_Float16)c2;
}

// s_acc init with cheb0 = I:  s_acc[g,n,o] = alpha0 + x*beta0[o] + wx*theta0[o]
__global__ void k_sacc_init(const float* __restrict__ x, const float* __restrict__ wx,
                            const float* __restrict__ alpha, const float* __restrict__ beta,
                            const float* __restrict__ theta, float* __restrict__ sacc) {
    long idx = (long)blockIdx.x * blockDim.x + threadIdx.x;
    if (idx >= (long)768 * NB * CO) return;
    int o = (int)(idx & 63);
    long gn = idx >> 6;                       // g*512+n
    float xv = x[gn], wv = wx[gn];
    sacc[idx] = alpha[0] + xv * beta[o] + wv * theta[o];
}

// s_r for cheb k, transposed to (g, o, n) in f16 for GEMM B-operand
__global__ void k_srT(const float* __restrict__ x, const float* __restrict__ wx,
                      const float* __restrict__ alpha, const float* __restrict__ beta,
                      const float* __restrict__ theta, _Float16* __restrict__ srT, int kk) {
    long idx = (long)blockIdx.x * blockDim.x + threadIdx.x;
    if (idx >= (long)768 * CO * NB) return;
    int n = (int)(idx & 511);
    int o = (int)((idx >> 9) & 63);
    long g = idx >> 15;
    long gn = g * NB + n;
    float v = alpha[kk] + x[gn] * beta[kk * 64 + o] + wx[gn] * theta[kk * 64 + o];
    srT[idx] = (_Float16)v;
}

// s_out[b,n,o,u] = sum_t relu(s_acc[b,t,n,o]) * slin_w[u,t] + slin_b[u]
__global__ void k_sout(const float* __restrict__ sacc, const float* __restrict__ sw,
                       const float* __restrict__ sb, float* __restrict__ sout) {
    long idx = (long)blockIdx.x * blockDim.x + threadIdx.x;
    if (idx >= (long)BB * NB * CO * TOUT) return;
    int u = (int)(idx % TOUT);
    long r = idx / TOUT;
    int o = (int)(r & 63);
    int n = (int)((r >> 6) & 511);
    int b = (int)(r >> 15);
    float acc = sb[u];
    for (int t = 0; t < TIN; ++t) {
        float v = sacc[(((long)(b * TIN + t) * NB + n) << 6) + o];
        acc += fmaxf(v, 0.f) * sw[u * TIN + t];
    }
    sout[idx] = acc;
}

// h init: h[s,t,d] = x[b,t,n] + sinusoidal PE(t,d); also f16 copy
__global__ void k_hinit(const float* __restrict__ x, float* __restrict__ h,
                        _Float16* __restrict__ hh) {
    long idx = (long)blockIdx.x * blockDim.x + threadIdx.x;
    if (idx >= ROWS * DM) return;
    int d = (int)(idx & 63);
    int t = (int)((idx >> 6) % TIN);
    long s = idx / (TIN * DM);
    int b = (int)(s >> 9), n = (int)(s & 511);
    int j = d >> 1;
    float div = __expf(-(float)(2 * j) * 0.14391156f);   // ln(10000)/64
    float sn, cs; __sincosf((float)t * div, &sn, &cs);
    float pe = (d & 1) ? cs : sn;
    float v = x[((long)(b * TIN + t) * NB + n)] + pe;
    h[idx] = v; hh[idx] = (_Float16)v;
}

// AutoCorrelation: per series, DFT(L=24) corr, channel-mean, top-3 + softmax,
// circular delay aggregation of V. One 64-thread block per series.
__global__ __launch_bounds__(64)
void k_corr(const float* __restrict__ Q, const float* __restrict__ Kb,
            const float* __restrict__ V, _Float16* __restrict__ attn) {
    const long s = blockIdx.x;
    const int c = threadIdx.x;                 // channel = h*8+e
    const long base = s * (long)TIN * DM + c;
    float qr[TIN], kr[TIN];
    for (int l = 0; l < TIN; ++l) { qr[l] = Q[base + l * DM]; kr[l] = Kb[base + l * DM]; }
    float pr[13], pi[13];
    const float w0 = -6.2831853f / 24.f;
    #pragma unroll
    for (int f = 0; f <= 12; ++f) {
        float qre = 0, qim = 0, kre = 0, kim = 0;
        for (int l = 0; l < TIN; ++l) {
            float sn, cs; __sincosf(w0 * (float)(f * l), &sn, &cs);
            qre += qr[l] * cs; qim += qr[l] * sn;
            kre += kr[l] * cs; kim += kr[l] * sn;
        }
        pr[f] = qre * kre + qim * kim;         // Q * conj(K)
        pi[f] = qim * kre - qre * kim;
    }
    __shared__ float mean[TIN];
    __shared__ float wts[3];
    __shared__ int   del[3];
    if (c < TIN) mean[c] = 0.f;
    __syncthreads();
    const float w1 = 6.2831853f / 24.f;
    for (int l = 0; l < TIN; ++l) {
        float acc = pr[0] + ((l & 1) ? -pr[12] : pr[12]);
        for (int f = 1; f <= 11; ++f) {
            float sn, cs; __sincosf(w1 * (float)(f * l), &sn, &cs);
            acc += 2.f * (pr[f] * cs - pi[f] * sn);
        }
        atomicAdd(&mean[l], acc * (1.f / 24.f) * (1.f / 64.f));
    }
    __syncthreads();
    if (c == 0) {                              // top_k = int(log(24)) = 3
        float m0 = -1e30f, m1 = -1e30f, m2 = -1e30f; int i0 = 0, i1 = 0, i2 = 0;
        for (int l = 0; l < TIN; ++l) {
            float v = mean[l];
            if (v > m0)      { m2 = m1; i2 = i1; m1 = m0; i1 = i0; m0 = v; i0 = l; }
            else if (v > m1) { m2 = m1; i2 = i1; m1 = v; i1 = l; }
            else if (v > m2) { m2 = v; i2 = l; }
        }
        float e0 = __expf(m0 - m0), e1 = __expf(m1 - m0), e2 = __expf(m2 - m0);
        float inv = 1.f / (e0 + e1 + e2);
        wts[0] = e0 * inv; wts[1] = e1 * inv; wts[2] = e2 * inv;
        del[0] = i0; del[1] = i1; del[2] = i2;
    }
    __syncthreads();
    for (int l = 0; l < TIN; ++l) qr[l] = V[base + l * DM];   // reuse regs for V
    for (int l = 0; l < TIN; ++l) {
        float acc = wts[0] * qr[(l + del[0]) % TIN]
                  + wts[1] * qr[(l + del[1]) % TIN]
                  + wts[2] * qr[(l + del[2]) % TIN];
        attn[base + l * DM] = (_Float16)acc;
    }
}

// h = (h + delta) - moving_avg_25(h + delta)  (edge-replicate pad), write f32 + f16
__global__ void k_decomp(float* __restrict__ h, const float* __restrict__ delta,
                         _Float16* __restrict__ hh) {
    long idx = (long)blockIdx.x * blockDim.x + threadIdx.x;
    if (idx >= (long)SEQ * DM) return;
    long s = idx >> 6; int d = (int)(idx & 63);
    long base = s * (long)TIN * DM + d;
    float y[TIN];
    for (int l = 0; l < TIN; ++l) y[l] = h[base + l * DM] + delta[base + l * DM];
    float sum = 12.f * y[0];
    for (int i = 0; i <= 12; ++i) sum += y[i];
    for (int l = 0; l < TIN; ++l) {
        float out = y[l] - sum * (1.f / 25.f);
        h[base + l * DM] = out;
        hh[base + l * DM] = (_Float16)out;
        sum += y[(l + 13 < TIN) ? l + 13 : TIN - 1] - y[(l >= 12) ? l - 12 : 0];
    }
}

// LayerNorm over d (per (s,l)) with gamma/beta -> hn
__global__ void k_ln(const float* __restrict__ h, const float* __restrict__ g,
                     const float* __restrict__ b, float* __restrict__ hn) {
    long idx = (long)blockIdx.x * blockDim.x + threadIdx.x;
    if (idx >= ROWS) return;
    long base = idx * DM;
    float mu = 0.f;
    for (int d = 0; d < DM; ++d) mu += h[base + d];
    mu *= (1.f / DM);
    float var = 0.f;
    for (int d = 0; d < DM; ++d) { float t = h[base + d] - mu; var += t * t; }
    float inv = rsqrtf(var * (1.f / DM) + 1e-5f);
    for (int d = 0; d < DM; ++d)
        hn[base + d] = (h[base + d] - mu) * inv * g[d] + b[d];
}

// subtract per-(s,d) mean over l (my_Layernorm), emit f16 flat rows for decoder
__global__ void k_submean(const float* __restrict__ hn, _Float16* __restrict__ hh) {
    long idx = (long)blockIdx.x * blockDim.x + threadIdx.x;
    if (idx >= (long)SEQ * DM) return;
    long s = idx >> 6; int d = (int)(idx & 63);
    long base = s * (long)TIN * DM + d;
    float m = 0.f;
    for (int l = 0; l < TIN; ++l) m += hn[base + l * DM];
    m *= (1.f / TIN);
    for (int l = 0; l < TIN; ++l) hh[base + l * DM] = (_Float16)(hn[base + l * DM] - m);
}

// out[b,t,n,o] = relu( s_out[b,n,o,t] * (t_out[b,n,t]*tlin_w[o] + tlin_b[o]) )
__global__ void k_final(const float* __restrict__ sout, const float* __restrict__ tout,
                        const float* __restrict__ tw, const float* __restrict__ tb,
                        float* __restrict__ out) {
    long idx = (long)blockIdx.x * blockDim.x + threadIdx.x;
    if (idx >= (long)BB * TOUT * NB * CO) return;
    int o = (int)(idx & 63);
    int n = (int)((idx >> 6) & 511);
    int t = (int)((idx >> 15) % TOUT);
    int b = (int)(idx / ((long)TOUT * NB * CO));
    long bn = (long)b * NB + n;
    float sv = sout[(bn * CO + o) * TOUT + t];
    float tv = tout[bn * TOUT + t] * tw[o] + tb[o];
    out[idx] = fmaxf(sv * tv, 0.f);
}

// =========================================================================
extern "C" void kernel_launch(void* const* d_in, const int* in_sizes, int n_in,
                              void* d_out, int out_size, void* d_ws, size_t ws_size,
                              hipStream_t stream) {
    const float* s_w    = (const float*)d_in[0];
    const float* x      = (const float*)d_in[1];
    const float* swnn_w = (const float*)d_in[2];
    const float* swnn_b = (const float*)d_in[3];
    const float* alpha  = (const float*)d_in[4];
    const float* beta   = (const float*)d_in[5];
    const float* theta  = (const float*)d_in[6];
    const float* wq = (const float*)d_in[7];  const float* bq = (const float*)d_in[8];
    const float* wk = (const float*)d_in[9];  const float* bk = (const float*)d_in[10];
    const float* wv = (const float*)d_in[11]; const float* bv = (const float*)d_in[12];
    const float* wo = (const float*)d_in[13]; const float* bo = (const float*)d_in[14];
    const float* c1w = (const float*)d_in[15]; const float* c1b = (const float*)d_in[16];
    const float* c2w = (const float*)d_in[17]; const float* c2b = (const float*)d_in[18];
    const float* ng  = (const float*)d_in[19]; const float* nb  = (const float*)d_in[20];
    const float* dw  = (const float*)d_in[21]; const float* db  = (const float*)d_in[22];
    const float* slw = (const float*)d_in[23]; const float* slb = (const float*)d_in[24];
    const float* tlw = (const float*)d_in[25]; const float* tlb = (const float*)d_in[26];
    float* out = (float*)d_out;

    char* ws = (char*)d_ws;
    size_t off = 0;
    auto alloc = [&](size_t bytes) -> char* {
        char* p = ws + off; off = (off + bytes + 255) & ~(size_t)255; return p;
    };

    _Float16* sw16  = (_Float16*)alloc((size_t)NB * NB * 2);
    _Float16* snn16 = (_Float16*)alloc((size_t)NB * NB * 2);
    float*    Wbuf  = (float*)   alloc((size_t)NB * NB * 4);
    float*    lam   = (float*)   alloc(256);
    _Float16* WT16  = (_Float16*)alloc((size_t)NB * NB * 2);
    _Float16* C1T   = (_Float16*)alloc((size_t)NB * NB * 2);
    _Float16* C2T   = (_Float16*)alloc((size_t)NB * NB * 2);
    _Float16* x16   = (_Float16*)alloc((size_t)768 * NB * 2);
    float*    wx    = (float*)   alloc((size_t)768 * NB * 4);
    _Float16* wq16  = (_Float16*)alloc(DM * DM * 2);
    _Float16* wk16  = (_Float16*)alloc(DM * DM * 2);
    _Float16* wv16  = (_Float16*)alloc(DM * DM * 2);
    _Float16* wo16  = (_Float16*)alloc(DM * DM * 2);
    _Float16* c1w16 = (_Float16*)alloc(DFF * DM * 2);
    _Float16* c2w16 = (_Float16*)alloc(DM * DFF * 2);
    _Float16* dec16 = (_Float16*)alloc((size_t)16 * (TIN * DM) * 2);   // 12 rows + zero pad
    float*    tout  = (float*)   alloc((size_t)SEQ * TOUT * 4);
    float*    sout  = (float*)   alloc((size_t)BB * NB * CO * TOUT * 4);
    _Float16* hh    = (_Float16*)alloc(ROWS * DM * 2);
    const size_t POOL = (size_t)768 * NB * CO * 4;                     // == ROWS*DM*4
    char* pool1 = alloc(POOL);                 // s_acc  -> h (f32)
    char* pool2 = alloc(POOL);                 // q
    char* pool3 = alloc(POOL);                 // s_rT   -> k
    char* pool4 = alloc(POOL);                 // v
    char* pool5 = alloc(POOL);                 // o / y2 / hn
    char* pool6 = alloc((size_t)ROWS * DFF * 2);  // attn16 / ffn-hidden f16
    float*    sacc   = (float*)pool1;   float* hbuf = (float*)pool1;
    float*    qb     = (float*)pool2;
    _Float16* srT    = (_Float16*)pool3; float* kb  = (float*)pool3;
    float*    vbuf   = (float*)pool4;
    float*    fbuf   = (float*)pool5;
    _Float16* attn16 = (_Float16*)pool6; _Float16* y16 = (_Float16*)pool6;

    auto cdiv = [](long n, long d) { return (int)((n + d - 1) / d); };
    auto gemm = [&](const _Float16* Aa, const _Float16* Bt, const float* bias,
                    float* Cf, _Float16* Ch, int M, int N, int K, int Nstore, int ldc,
                    long sA, long sB, long sC, int batches, int relu, int acc) {
        int tilesN = N >> 4;
        if ((tilesN & 3) == 0) {
            int strips = (M >> 4) * (tilesN >> 2);
            dim3 grid(cdiv(strips, 4), batches);
            k_gemm<4><<<grid, 128, 0, stream>>>(Aa, Bt, bias, Cf, Ch, M, K, Nstore, ldc,
                                                sA, sB, sC, tilesN, relu, acc);
        } else {
            int strips = (M >> 4) * tilesN;
            dim3 grid(cdiv(strips, 4), batches);
            k_gemm<1><<<grid, 128, 0, stream>>>(Aa, Bt, bias, Cf, Ch, M, K, Nstore, ldc,
                                                sA, sB, sC, tilesN, relu, acc);
        }
    };

    // ---- operand conversions to f16 ----
    k_cvt<<<cdiv((long)NB*NB,256),256,0,stream>>>(s_w,    sw16,  (long)NB*NB);
    k_cvt<<<cdiv((long)NB*NB,256),256,0,stream>>>(swnn_w, snn16, (long)NB*NB);
    k_cvt<<<cdiv((long)768*NB,256),256,0,stream>>>(x,     x16,   (long)768*NB);
    k_cvt<<<cdiv(DM*DM,256),256,0,stream>>>(wq, wq16, DM*DM);
    k_cvt<<<cdiv(DM*DM,256),256,0,stream>>>(wk, wk16, DM*DM);
    k_cvt<<<cdiv(DM*DM,256),256,0,stream>>>(wv, wv16, DM*DM);
    k_cvt<<<cdiv(DM*DM,256),256,0,stream>>>(wo, wo16, DM*DM);
    k_cvt<<<cdiv(DFF*DM,256),256,0,stream>>>(c1w, c1w16, DFF*DM);
    k_cvt<<<cdiv(DM*DFF,256),256,0,stream>>>(c2w, c2w16, DM*DFF);
    k_fill0h<<<cdiv(16L*TIN*DM,256),256,0,stream>>>(dec16, 16L*TIN*DM);
    k_cvt<<<cdiv((long)TOUT*TIN*DM,256),256,0,stream>>>(dw, dec16, (long)TOUT*TIN*DM);

    // ---- spatial branch ----
    gemm(sw16, snn16, swnn_b, Wbuf, nullptr, NB, NB, NB, NB, NB, 0, 0, 0, 1, 0, 0);
    k_softmax512<<<NB, 256, 0, stream>>>(Wbuf);
    k_power<<<1, NB, 0, stream>>>(Wbuf, lam);
    k_build<<<cdiv((long)NB*NB,256),256,0,stream>>>(Wbuf, lam, WT16, C1T, C2T);
    gemm(x16, WT16, nullptr, wx, nullptr, 768, NB, NB, NB, NB, 0, 0, 0, 1, 0, 0);
    k_sacc_init<<<cdiv((long)768*NB*CO,256),256,0,stream>>>(x, wx, alpha, beta, theta, sacc);
    const _Float16* chebT[2] = { C1T, C2T };
    for (int kk = 1; kk <= 2; ++kk) {
        k_srT<<<cdiv((long)768*CO*NB,256),256,0,stream>>>(x, wx, alpha, beta, theta, srT, kk);
        gemm(chebT[kk-1], srT, nullptr, sacc, nullptr, NB, CO, NB, CO, CO,
             0, (long)CO*NB, (long)NB*CO, 768, 0, 1);
    }
    k_sout<<<cdiv((long)BB*NB*CO*TOUT,256),256,0,stream>>>(sacc, slw, slb, sout);

    // ---- temporal branch ----
    k_hinit<<<cdiv(ROWS*DM,256),256,0,stream>>>(x, hbuf, hh);
    for (int layer = 0; layer < 2; ++layer) {
        gemm(hh, wq16, bq, qb,   nullptr, (int)ROWS, DM, DM, DM, DM, 0, 0, 0, 1, 0, 0);
        gemm(hh, wk16, bk, kb,   nullptr, (int)ROWS, DM, DM, DM, DM, 0, 0, 0, 1, 0, 0);
        gemm(hh, wv16, bv, vbuf, nullptr, (int)ROWS, DM, DM, DM, DM, 0, 0, 0, 1, 0, 0);
        k_corr<<<SEQ, 64, 0, stream>>>(qb, kb, vbuf, attn16);
        gemm(attn16, wo16, bo, fbuf, nullptr, (int)ROWS, DM, DM, DM, DM, 0, 0, 0, 1, 0, 0);
        k_decomp<<<cdiv((long)SEQ*DM,256),256,0,stream>>>(hbuf, fbuf, hh);
        gemm(hh, c1w16, c1b, nullptr, y16, (int)ROWS, DFF, DM, DFF, DFF, 0, 0, 0, 1, 1, 0);
        gemm(y16, c2w16, c2b, fbuf, nullptr, (int)ROWS, DM, DFF, DM, DM, 0, 0, 0, 1, 0, 0);
        k_decomp<<<cdiv((long)SEQ*DM,256),256,0,stream>>>(hbuf, fbuf, hh);
    }
    k_ln<<<cdiv(ROWS,256),256,0,stream>>>(hbuf, ng, nb, fbuf);
    k_submean<<<cdiv((long)SEQ*DM,256),256,0,stream>>>(fbuf, hh);
    gemm(hh, dec16, db, tout, nullptr, SEQ, 16, TIN*DM, TOUT, TOUT, 0, 0, 0, 1, 1, 0);

    // ---- combine ----
    k_final<<<cdiv((long)BB*TOUT*NB*CO,256),256,0,stream>>>(sout, tout, tlw, tlb, out);
}
